// BipartiteGNN_69922067579512
// MI455X (gfx1250) — matrix operand order repaired and checked by reading, not compiled, not run on hardware
//
#include <hip/hip_runtime.h>

#define D 128
#define VOCAB 2000
#define NUM_ITERS 6
#define NUM_V 98304
#define NUM_E (NUM_V / 3)

typedef __attribute__((ext_vector_type(16))) __bf16 bf16x16;
typedef __attribute__((ext_vector_type(8))) float f32x8;

struct FragU { uint4 q[2]; };

// ---------- helpers ----------------------------------------------------------

__device__ __forceinline__ __bf16 f2bf(float f) {
    union { float f; unsigned u; } x; x.f = f;
    unsigned r = x.u + 0x7FFFu + ((x.u >> 16) & 1u);   // round-to-nearest-even
    unsigned short h = (unsigned short)(r >> 16);
    return __builtin_bit_cast(__bf16, h);
}

__device__ __forceinline__ float sigmoidf(float x) {
    return 1.0f / (1.0f + __expf(-x));
}

// A fragment (16x32, 16-bit): lane l holds row (l&15);
// halves[0..7] = K[k0 + (l>>4)*8 ..), halves[8..15] = K[k0+16 + (l>>4)*8 ..).
// Works for both global and LDS source pointers (compiler picks ds_load/global_load).
__device__ __forceinline__ bf16x16 load_frag_a(const __bf16* base, int ld, int row0, int k0) {
    const int lane = threadIdx.x & 31;
    const __bf16* p = base + (size_t)(row0 + (lane & 15)) * ld + k0 + ((lane >> 4) << 3);
    FragU f;
    f.q[0] = *(const uint4*)(p);
    f.q[1] = *(const uint4*)(p + 16);
    return __builtin_bit_cast(bf16x16, f);
}

// B fragment (32x16, 16-bit), B = W^T with W row-major [N x K]:
// lane l holds column n=(l&15); halves[0..7] = K[k0 + (l>>4)*8 ..), halves[8..15] = K[k0+16 + (l>>4)*8 ..).
__device__ __forceinline__ bf16x16 load_frag_b(const __bf16* base, int ld, int row0, int k0) {
    const int lane = threadIdx.x & 31;
    const __bf16* p = base + (size_t)(row0 + (lane & 15)) * ld + k0 + ((lane >> 4) << 3);
    FragU f;
    f.q[0] = *(const uint4*)(p);
    f.q[1] = *(const uint4*)(p + 16);
    return __builtin_bit_cast(bf16x16, f);
}

__device__ __forceinline__ bf16x16 load_frag_b_clamp(const __bf16* base, int ld, int row0,
                                                     int k0, int maxrow) {
    const int lane = threadIdx.x & 31;
    int r = row0 + (lane & 15);
    if (r >= maxrow) r = maxrow - 1;
    const __bf16* p = base + (size_t)r * ld + k0 + ((lane >> 4) << 3);
    FragU f;
    f.q[0] = *(const uint4*)(p);
    f.q[1] = *(const uint4*)(p + 16);
    return __builtin_bit_cast(bf16x16, f);
}

__device__ __forceinline__ f32x8 wmma_bf16(bf16x16 a, bf16x16 b, f32x8 c) {
    return __builtin_amdgcn_wmma_f32_16x16x32_bf16(false, a, false, b, (short)0, c,
                                                   false, false);
}

// C/D layout: lane l, VGPR r -> (m = r + 8*(l>>4), n = l&15)
__device__ __forceinline__ void store_acc_lds(float* g, int ldg, int m0, int n0, f32x8 acc) {
    const int lane = threadIdx.x & 31;
    const int m = m0 + ((lane >> 4) << 3);
    const int n = n0 + (lane & 15);
#pragma unroll
    for (int r = 0; r < 8; ++r) g[(m + r) * ldg + n] = acc[r];
}

// ---------- CDNA5 async global->LDS (ASYNCcnt path) --------------------------
// Generic pointers to __shared__ carry the wave-relative LDS byte address in
// their low 32 bits (ISA 10.2 aperture rule), which is exactly what the
// instruction's VDST operand wants.
__device__ __forceinline__ void async_g2l_b128(void* lds_dst, const void* gsrc) {
    unsigned lds_addr = (unsigned)(unsigned long long)lds_dst;
    unsigned long long gaddr = (unsigned long long)gsrc;
    asm volatile("global_load_async_to_lds_b128 %0, %1, off"
                 :: "v"(lds_addr), "v"(gaddr)
                 : "memory");
}

__device__ __forceinline__ void wait_async0() {
    asm volatile("s_wait_asynccnt 0" ::: "memory");
}

// ---------- init / pack ------------------------------------------------------

__global__ __launch_bounds__(256) void init_hv_kernel(const int* __restrict__ xv,
                                                      const float* __restrict__ emb,
                                                      __bf16* __restrict__ hv,
                                                      float* __restrict__ cv) {
    int idx = blockIdx.x * 256 + threadIdx.x;
    if (idx >= NUM_V * D) return;
    int v = idx >> 7, d = idx & 127;
    hv[idx] = f2bf(emb[(size_t)xv[v] * D + d]);
    cv[idx] = 0.0f;
}

__global__ __launch_bounds__(256) void init_he_kernel(const float* __restrict__ w,
                                                      const float* __restrict__ b,
                                                      __bf16* __restrict__ he,
                                                      float* __restrict__ ce) {
    int idx = blockIdx.x * 256 + threadIdx.x;
    if (idx >= NUM_E * D) return;
    int d = idx & 127;
    he[idx] = f2bf(w[d] + b[d]);   // edge_init_w is [D,1] -> w[d]; Linear on ones
    ce[idx] = 0.0f;
}

__global__ __launch_bounds__(256) void pack_weights_kernel(
    const float* __restrict__ Wih_v2e, const float* __restrict__ Whh_v2e,
    const float* __restrict__ bih_v2e, const float* __restrict__ bhh_v2e,
    const float* __restrict__ Wih_e2v, const float* __restrict__ Whh_e2v,
    const float* __restrict__ bih_e2v, const float* __restrict__ bhh_e2v,
    const float* __restrict__ out_w,
    __bf16* __restrict__ Wv2e,   // [512][512]   cols: 0..383 = h_v, 384..511 = h_e
    __bf16* __restrict__ We2v,   // [3][512][256] cols: 0..127 = h_e, 128..255 = h_v_p
    __bf16* __restrict__ Wo,     // [2000][128]
    float* __restrict__ be,      // [512]
    float* __restrict__ bv)      // [3][512]
{
    int idx = blockIdx.x * 256 + threadIdx.x;
    if (idx < 512 * 512) {
        int n = idx >> 9, k = idx & 511;
        float v = (k < 384) ? Wih_v2e[n * 384 + k] : Whh_v2e[n * 128 + (k - 384)];
        Wv2e[idx] = f2bf(v);
        return;
    }
    idx -= 512 * 512;
    if (idx < 3 * 512 * 256) {
        int p = idx / (512 * 256), rem = idx % (512 * 256);
        int n = rem >> 8, k = rem & 255;
        size_t row = ((size_t)p * 512 + n) * 128;
        float v = (k < 128) ? Wih_e2v[row + k] : Whh_e2v[row + (k - 128)];
        We2v[idx] = f2bf(v);
        return;
    }
    idx -= 3 * 512 * 256;
    if (idx < VOCAB * D) { Wo[idx] = f2bf(out_w[idx]); return; }
    idx -= VOCAB * D;
    if (idx < 512) { be[idx] = bih_v2e[idx] + bhh_v2e[idx]; return; }
    idx -= 512;
    if (idx < 3 * 512) { bv[idx] = bih_e2v[idx] + bhh_e2v[idx]; return; }
}

// ---------- vertex -> edge: gates = [h_v | h_e] @ Wv2e^T, fused LSTM ---------
// A tile (32 rows x 512 K, concatenated) is async-staged into LDS once per
// block; all 8 waves read fragments from LDS (kills the 8x redundant VMEM).

#define V2E_LDA 528   // 512 + 16 elem pad -> row stride 1056 B (8-bank rotation)

__global__ __launch_bounds__(256, 1) void v2e_kernel(const __bf16* __restrict__ hv,
                                                     __bf16* __restrict__ he,
                                                     float* __restrict__ ce,
                                                     const __bf16* __restrict__ W,
                                                     const float* __restrict__ be) {
    __shared__ float g[32 * 512];                   // 64 KB gate staging
    __shared__ __bf16 shA[32 * V2E_LDA];            // 33 KB A tile
    const int r0 = blockIdx.x * 32;
    const int wn = (threadIdx.x >> 5) * 64;
    const int tid = threadIdx.x;

    // async stage: h_v rows (768 B each) -> LDS cols [0,384); h_e (256 B) -> [384,512)
#pragma unroll
    for (int i = 0; i < 6; ++i) {                   // 1536 chunks of 16 B
        int chunk = tid + i * 256;
        int row = chunk / 48, c = chunk % 48;
        async_g2l_b128((char*)shA + row * (V2E_LDA * 2) + c * 16,
                       (const char*)hv + (size_t)(r0 + row) * 768 + c * 16);
    }
#pragma unroll
    for (int i = 0; i < 2; ++i) {                   // 512 chunks of 16 B
        int chunk = tid + i * 256;
        int row = chunk / 16, c = chunk % 16;
        async_g2l_b128((char*)shA + row * (V2E_LDA * 2) + 768 + c * 16,
                       (const char*)he + (size_t)(r0 + row) * 256 + c * 16);
    }
    wait_async0();
    __syncthreads();

    f32x8 acc[2][4] = {};
    for (int kt = 0; kt < 16; ++kt) {               // K = 512, concatenated in LDS
        bf16x16 a0 = load_frag_a(shA, V2E_LDA, 0, kt * 32);
        bf16x16 a1 = load_frag_a(shA, V2E_LDA, 16, kt * 32);
#pragma unroll
        for (int j = 0; j < 4; ++j) {
            bf16x16 b = load_frag_b(W, 512, wn + j * 16, kt * 32);
            acc[0][j] = wmma_bf16(a0, b, acc[0][j]);
            acc[1][j] = wmma_bf16(a1, b, acc[1][j]);
        }
    }
#pragma unroll
    for (int i = 0; i < 2; ++i)
#pragma unroll
        for (int j = 0; j < 4; ++j)
            store_acc_lds(g, 512, i * 16, wn + j * 16, acc[i][j]);
    __syncthreads();

    // LSTM cell epilogue: 32 rows x 128 hidden units
#pragma unroll
    for (int t = 0; t < 16; ++t) {
        int idx = threadIdx.x + t * 256;            // 0..4095
        int m = idx >> 7, j = idx & 127;
        float gi = g[m * 512 + j]        + be[j];
        float gf = g[m * 512 + 128 + j]  + be[128 + j];
        float gg = g[m * 512 + 256 + j]  + be[256 + j];
        float go = g[m * 512 + 384 + j]  + be[384 + j];
        size_t off = (size_t)(r0 + m) * 128 + j;
        float c_new = sigmoidf(gf) * ce[off] + sigmoidf(gi) * tanhf(gg);
        ce[off] = c_new;
        he[off] = f2bf(sigmoidf(go) * tanhf(c_new));
    }
}

// ---------- edge -> vertex (per position p): fused masked LSTM ---------------

#define E2V_LDA 272   // 256 + 16 elem pad -> row stride 544 B

__global__ __launch_bounds__(256, 1) void e2v_kernel(__bf16* __restrict__ hv,
                                                     float* __restrict__ cv,
                                                     const __bf16* __restrict__ he,
                                                     const __bf16* __restrict__ W,
                                                     const float* __restrict__ bv,
                                                     const int* __restrict__ mask) {
    __shared__ float g[32 * 512];
    __shared__ __bf16 shA[32 * E2V_LDA];            // 17 KB A tile
    const int p  = blockIdx.y;
    const int r0 = blockIdx.x * 32;
    const int wn = (threadIdx.x >> 5) * 64;
    const int tid = threadIdx.x;
    const __bf16* Wp = W + (size_t)p * 512 * 256;
    const float*  bp = bv + p * 512;

    // async stage: h_e rows (256 B) -> LDS cols [0,128); h_v[:,p,:] -> [128,256)
#pragma unroll
    for (int i = 0; i < 2; ++i) {                   // 512 chunks
        int chunk = tid + i * 256;
        int row = chunk / 16, c = chunk % 16;
        async_g2l_b128((char*)shA + row * (E2V_LDA * 2) + c * 16,
                       (const char*)he + (size_t)(r0 + row) * 256 + c * 16);
    }
#pragma unroll
    for (int i = 0; i < 2; ++i) {                   // 512 chunks
        int chunk = tid + i * 256;
        int row = chunk / 16, c = chunk % 16;
        async_g2l_b128((char*)shA + row * (E2V_LDA * 2) + 256 + c * 16,
                       (const char*)hv + ((size_t)(r0 + row) * 3 + p) * 256 + c * 16);
    }
    wait_async0();
    __syncthreads();

    f32x8 acc[2][4] = {};
    for (int kt = 0; kt < 8; ++kt) {                // K = 256, concatenated in LDS
        bf16x16 a0 = load_frag_a(shA, E2V_LDA, 0, kt * 32);
        bf16x16 a1 = load_frag_a(shA, E2V_LDA, 16, kt * 32);
#pragma unroll
        for (int j = 0; j < 4; ++j) {
            bf16x16 b = load_frag_b(Wp, 256, wn + j * 16, kt * 32);
            acc[0][j] = wmma_bf16(a0, b, acc[0][j]);
            acc[1][j] = wmma_bf16(a1, b, acc[1][j]);
        }
    }
#pragma unroll
    for (int i = 0; i < 2; ++i)
#pragma unroll
        for (int j = 0; j < 4; ++j)
            store_acc_lds(g, 512, i * 16, wn + j * 16, acc[i][j]);
    __syncthreads();

#pragma unroll
    for (int t = 0; t < 16; ++t) {
        int idx = threadIdx.x + t * 256;
        int m = idx >> 7, j = idx & 127;
        int vert = (r0 + m) * 3 + p;
        if (mask[vert] != 0) {
            float gi = g[m * 512 + j]        + bp[j];
            float gf = g[m * 512 + 128 + j]  + bp[128 + j];
            float gg = g[m * 512 + 256 + j]  + bp[256 + j];
            float go = g[m * 512 + 384 + j]  + bp[384 + j];
            size_t off = (size_t)vert * 128 + j;
            float c_new = sigmoidf(gf) * cv[off] + sigmoidf(gi) * tanhf(gg);
            cv[off] = c_new;
            hv[off] = f2bf(sigmoidf(go) * tanhf(c_new));
        }
    }
}

// ---------- output projection: logits = h_v @ out_w^T + out_b ----------------

__global__ __launch_bounds__(256, 1) void out_kernel(const __bf16* __restrict__ hv,
                                                     const __bf16* __restrict__ Wo,
                                                     const float* __restrict__ bo,
                                                     float* __restrict__ out) {
    const int wid = threadIdx.x >> 5;
    const int m0 = blockIdx.y * 64 + (wid & 1) * 32;
    const int n0 = blockIdx.x * 128 + (wid >> 1) * 32;

    f32x8 acc[2][2] = {};
    for (int kt = 0; kt < 4; ++kt) {                // K = 128
        bf16x16 a0 = load_frag_a(hv, 128, m0, kt * 32);
        bf16x16 a1 = load_frag_a(hv, 128, m0 + 16, kt * 32);
        bf16x16 b0 = load_frag_b_clamp(Wo, 128, n0, kt * 32, VOCAB);
        bf16x16 b1 = load_frag_b_clamp(Wo, 128, n0 + 16, kt * 32, VOCAB);
        acc[0][0] = wmma_bf16(a0, b0, acc[0][0]);
        acc[0][1] = wmma_bf16(a0, b1, acc[0][1]);
        acc[1][0] = wmma_bf16(a1, b0, acc[1][0]);
        acc[1][1] = wmma_bf16(a1, b1, acc[1][1]);
    }
    const int lane = threadIdx.x & 31;
#pragma unroll
    for (int i = 0; i < 2; ++i)
#pragma unroll
        for (int j = 0; j < 2; ++j) {
            int m = m0 + i * 16 + ((lane >> 4) << 3);
            int n = n0 + j * 16 + (lane & 15);
            if (n < VOCAB) {
                float bn = bo[n];
#pragma unroll
                for (int r = 0; r < 8; ++r)
                    out[(size_t)(m + r) * VOCAB + n] = acc[i][j][r] + bn;
            }
        }
}

// ---------- host orchestration ----------------------------------------------

extern "C" void kernel_launch(void* const* d_in, const int* in_sizes, int n_in,
                              void* d_out, int out_size, void* d_ws, size_t ws_size,
                              hipStream_t stream) {
    const int*   x_v       = (const int*)  d_in[0];
    const int*   tmask     = (const int*)  d_in[1];
    const float* emb       = (const float*)d_in[2];
    const float* ei_w      = (const float*)d_in[3];
    const float* ei_b      = (const float*)d_in[4];
    const float* Wih_v2e   = (const float*)d_in[5];
    const float* Whh_v2e   = (const float*)d_in[6];
    const float* bih_v2e   = (const float*)d_in[7];
    const float* bhh_v2e   = (const float*)d_in[8];
    const float* Wih_e2v   = (const float*)d_in[9];
    const float* Whh_e2v   = (const float*)d_in[10];
    const float* bih_e2v   = (const float*)d_in[11];
    const float* bhh_e2v   = (const float*)d_in[12];
    const float* out_w     = (const float*)d_in[13];
    const float* out_b     = (const float*)d_in[14];

    // carve workspace (256B-aligned slabs)
    char* w = (char*)d_ws;
    auto carve = [&](size_t bytes) {
        void* p = (void*)w;
        w += (bytes + 255) & ~(size_t)255;
        return p;
    };
    __bf16* hv   = (__bf16*)carve((size_t)NUM_V * D * 2);
    float*  cv   = (float*) carve((size_t)NUM_V * D * 4);
    __bf16* he   = (__bf16*)carve((size_t)NUM_E * D * 2);
    float*  ce   = (float*) carve((size_t)NUM_E * D * 4);
    __bf16* Wv2e = (__bf16*)carve((size_t)512 * 512 * 2);
    __bf16* We2v = (__bf16*)carve((size_t)3 * 512 * 256 * 2);
    __bf16* Wo   = (__bf16*)carve((size_t)VOCAB * D * 2);
    float*  be   = (float*) carve((size_t)512 * 4);
    float*  bv   = (float*) carve((size_t)3 * 512 * 4);

    // init state + pack weights (bf16) every call (deterministic)
    init_hv_kernel<<<(NUM_V * D + 255) / 256, 256, 0, stream>>>(x_v, emb, hv, cv);
    init_he_kernel<<<(NUM_E * D + 255) / 256, 256, 0, stream>>>(ei_w, ei_b, he, ce);
    {
        int total = 512 * 512 + 3 * 512 * 256 + VOCAB * D + 512 + 3 * 512;
        pack_weights_kernel<<<(total + 255) / 256, 256, 0, stream>>>(
            Wih_v2e, Whh_v2e, bih_v2e, bhh_v2e,
            Wih_e2v, Whh_e2v, bih_e2v, bhh_e2v,
            out_w, Wv2e, We2v, Wo, be, bv);
    }

    for (int it = 0; it < NUM_ITERS; ++it) {
        v2e_kernel<<<NUM_E / 32, 256, 0, stream>>>(hv, he, ce, Wv2e, be);
        e2v_kernel<<<dim3(NUM_E / 32, 3), 256, 0, stream>>>(hv, cv, he, We2v, bv, tmask);
    }

    out_kernel<<<dim3((VOCAB + 127) / 128, NUM_V / 64), 256, 0, stream>>>(
        hv, Wo, out_b, (float*)d_out);
}
